// SCLoss_24678882083212
// MI455X (gfx1250) — matrix operand order repaired and checked by reading, not compiled, not run on hardware
//
#include <hip/hip_runtime.h>

// ---------------- types ----------------
typedef __attribute__((ext_vector_type(16))) __bf16 v16bf;
typedef __attribute__((ext_vector_type(8)))  __bf16 v8bf;
typedef __attribute__((ext_vector_type(8)))  float  v8f;

#define N_ROWS 8192
#define DIMK   256
#define INV_T  3.33333333333333f   // 1/0.3
#define JSPLIT 16
#define ROWBLK 128                 // 8192 / 64 rows per block (4 waves x 16)
#define JCOLS  (N_ROWS / JSPLIT)   // 512 columns per split
#define TILES  (JCOLS / 16)        // 32 column tiles per split
// LDS B-tile row stride: 256 bf16 + 8 pad = 264 elems = 528B = 132 dwords (== 4 mod 64 -> conflict-free)
#define BSTRIDE 264
#define BUFELEM (16 * BSTRIDE)     // elems per buffer
#define BUFBYTE (BUFELEM * 2)      // 8448 bytes per buffer

// round-to-nearest-even f32 -> bf16 bit pattern
static __device__ __forceinline__ unsigned short f2bf(float x) {
    unsigned int u = __float_as_uint(x);
    u += 0x7FFFu + ((u >> 16) & 1u);
    return (unsigned short)(u >> 16);
}

// ---------------- kernel 0: zero accumulators + output ----------------
__global__ void scl_init(float* acc, float* out) {
    int i = blockIdx.x * 256 + threadIdx.x;
    if (i < 3 * N_ROWS) acc[i] = 0.0f;
    if (i == 0) out[0] = 0.0f;
}

// ---------------- kernel 1: L2-normalize rows, store bf16 ----------------
__global__ void scl_norm(const float* __restrict__ f, unsigned short* __restrict__ fb) {
    int wave = threadIdx.x >> 5;
    int lane = threadIdx.x & 31;
    int row  = blockIdx.x * 8 + wave;
    const float* src = f + (size_t)row * DIMK;

    float x[8];
    float ss = 0.0f;
#pragma unroll
    for (int q = 0; q < 8; ++q) {
        x[q] = src[lane + 32 * q];
        ss += x[q] * x[q];
    }
#pragma unroll
    for (int m = 1; m < 32; m <<= 1) ss += __shfl_xor(ss, m, 32);

    float inv = 1.0f / fmaxf(sqrtf(ss), 1e-12f);
    unsigned short* dst = fb + (size_t)row * DIMK;
#pragma unroll
    for (int q = 0; q < 8; ++q) dst[lane + 32 * q] = f2bf(x[q] * inv);
}

// ---------------- kernel 2: fused Gram + reductions ----------------
// block = 4 waves (128 thr). wave owns 16 rows. blockIdx = rb + ROWBLK*js.
// B tiles double-buffered in LDS, filled with global_load_async_to_lds_b128.
__global__ void __launch_bounds__(128)
scl_main(const unsigned short* __restrict__ fbu,
         const long long* __restrict__ labels,
         float* __restrict__ acc) {
    __shared__ __align__(16) __bf16 Bbuf[2][BUFELEM];
    __shared__ int Lab[JCOLS];

    const __bf16* fb = (const __bf16*)fbu;
    float* sumexp = acc;
    float* possum = acc + N_ROWS;
    float* poscnt = acc + 2 * N_ROWS;

    int tid  = threadIdx.x;
    int wave = tid >> 5;
    int lane = tid & 31;
    int half = lane >> 4;
    int n    = lane & 15;

    int rb = blockIdx.x % ROWBLK;
    int js = blockIdx.x / ROWBLK;
    int i0 = rb * 64 + wave * 16;                 // this wave's 16-row strip

    // ---- preload this split's 512 column labels into LDS ----
    for (int q = tid; q < JCOLS; q += 128) Lab[q] = (int)labels[js * JCOLS + q];

    // ---- load A fragments once: K = 0..255 in 8 chunks of 32 ----
    v16bf A[8];
    const __bf16* arow = fb + (size_t)(i0 + n) * DIMK;
#pragma unroll
    for (int c = 0; c < 8; ++c) {
        v8bf lo = *(const v8bf*)(arow + c * 32 + 8 * half);
        v8bf hi = *(const v8bf*)(arow + c * 32 + 16 + 8 * half);
        A[c] = __builtin_shufflevector(lo, hi, 0,1,2,3,4,5,6,7,8,9,10,11,12,13,14,15);
    }

    int labi[8];
#pragma unroll
    for (int v = 0; v < 8; ++v) labi[v] = (int)labels[i0 + v + 8 * half];

    float accS[8], accP[8], accC[8];
#pragma unroll
    for (int v = 0; v < 8; ++v) { accS[v] = 0.f; accP[v] = 0.f; accC[v] = 0.f; }

    // ---- per-thread fixed copy offsets: 128 thr x 4 x 16B = 8KB tile ----
    unsigned loff[4];
    unsigned gofs[4];
#pragma unroll
    for (int k = 0; k < 4; ++k) {
        int q = tid + 128 * k;                 // chunk id 0..511
        int row = q >> 5, c16 = q & 31;        // source row, 16B chunk in row
        loff[k] = (unsigned)(uintptr_t)(&Bbuf[0][0]) + row * (BSTRIDE * 2) + c16 * 16;
        gofs[k] = row * (DIMK * 2) + c16 * 16;
    }
    const char* gtile0 = (const char*)(fb + (size_t)(js * JCOLS) * DIMK);

    // prologue: async-copy tile 0 into buffer 0
    {
        const char* g = gtile0;
#pragma unroll
        for (int k = 0; k < 4; ++k) {
            asm volatile("global_load_async_to_lds_b128 %0, %1, off"
                         :: "v"(loff[k]), "v"(g + gofs[k]) : "memory");
        }
    }

    for (int t = 0; t < TILES; ++t) {
        int bc = t & 1;
        int j0 = js * JCOLS + t * 16;

        asm volatile("s_wait_asynccnt 0" ::: "memory");  // own async copies landed
        __syncthreads();   // everyone's copies done; previous buffer's readers done

        if (t + 1 < TILES) {                   // prefetch tile t+1 into other buffer
            const char* g = gtile0 + (size_t)(t + 1) * 16 * (DIMK * 2);
            unsigned lb = (unsigned)(bc ^ 1) * BUFBYTE;
#pragma unroll
            for (int k = 0; k < 4; ++k) {
                asm volatile("global_load_async_to_lds_b128 %0, %1, off"
                             :: "v"(loff[k] + lb), "v"(g + gofs[k]) : "memory");
            }
        }

        // ---- load all 8 B fragments from LDS, then 8 back-to-back WMMAs ----
        const __bf16* bl = &Bbuf[bc][0] + n * BSTRIDE + 8 * half;
        v16bf B[8];
#pragma unroll
        for (int c = 0; c < 8; ++c) {
            v8bf blo = *(const v8bf*)(bl + c * 32);
            v8bf bhi = *(const v8bf*)(bl + c * 32 + 16);
            B[c] = __builtin_shufflevector(blo, bhi, 0,1,2,3,4,5,6,7,8,9,10,11,12,13,14,15);
        }

        v8f C = {};
#pragma unroll
        for (int c = 0; c < 8; ++c)
            C = __builtin_amdgcn_wmma_f32_16x16x32_bf16(
                    false, A[c], false, B[c], (short)0, C, false, false);

        int labj = Lab[t * 16 + n];

        // epilogue: C[v] = S'[row = i0+v+8h][col = j0+n] (pre-temperature)
#pragma unroll
        for (int v = 0; v < 8; ++v) {
            float val = C[v] * INV_T;
            int r = i0 + v + 8 * half;
            bool self = (j0 + n) == r;
            float e = self ? 0.0f : __expf(val);
            accS[v] += e;
            bool pos = (!self) && (labj == labi[v]);
            accP[v] += pos ? val : 0.0f;
            accC[v] += pos ? 1.0f : 0.0f;
        }
    }

    // reduce across the 16 column-lanes within each half (xor 1,2,4,8 stays in-half)
#pragma unroll
    for (int v = 0; v < 8; ++v) {
#pragma unroll
        for (int m = 1; m < 16; m <<= 1) {
            accS[v] += __shfl_xor(accS[v], m, 32);
            accP[v] += __shfl_xor(accP[v], m, 32);
            accC[v] += __shfl_xor(accC[v], m, 32);
        }
    }

    // lane with n==v commits row r = i0 + v + 8*half (16 lanes cover 16 rows)
#pragma unroll
    for (int v = 0; v < 8; ++v) {
        if (n == v) {
            int r = i0 + v + 8 * half;
            atomicAdd(&sumexp[r], accS[v]);
            atomicAdd(&possum[r], accP[v]);
            atomicAdd(&poscnt[r], accC[v]);
        }
    }
}

// ---------------- kernel 3: finalize scalar loss ----------------
__global__ void scl_final(const float* __restrict__ acc, float* __restrict__ out) {
    int r = blockIdx.x * 256 + threadIdx.x;
    const float* sumexp = acc;
    const float* possum = acc + N_ROWS;
    const float* poscnt = acc + 2 * N_ROWS;

    float k = poscnt[r];
    float li = possum[r] / fmaxf(k, 1.0f) - logf(sumexp[r]);
    float c = (k > 0.0f) ? -li : 0.0f;

#pragma unroll
    for (int m = 1; m < 32; m <<= 1) c += __shfl_xor(c, m, 32);

    __shared__ float ws8[8];
    int wave = threadIdx.x >> 5;
    int lane = threadIdx.x & 31;
    if (lane == 0) ws8[wave] = c;
    __syncthreads();
    if (threadIdx.x == 0) {
        float t = 0.0f;
#pragma unroll
        for (int w = 0; w < 8; ++w) t += ws8[w];
        atomicAdd(out, t);
    }
}

// ---------------- launcher ----------------
extern "C" void kernel_launch(void* const* d_in, const int* in_sizes, int n_in,
                              void* d_out, int out_size, void* d_ws, size_t ws_size,
                              hipStream_t stream) {
    const float*     features = (const float*)d_in[0];     // [8192, 256] f32
    const long long* labels   = (const long long*)d_in[1]; // [8192] i64
    float* out = (float*)d_out;                            // scalar f32

    // ws layout: [3*8192 f32 accumulators][8192*256 bf16 normalized features]
    float* acc = (float*)d_ws;
    unsigned short* fb = (unsigned short*)(acc + 3 * N_ROWS);

    scl_init <<<96,   256, 0, stream>>>(acc, out);
    scl_norm <<<N_ROWS / 8, 256, 0, stream>>>(features, fb);
    scl_main <<<ROWBLK * JSPLIT, 128, 0, stream>>>(fb, labels, acc);
    scl_final<<<N_ROWS / 256, 256, 0, stream>>>(acc, out);
}
